// DynamicGraphAdjacentMatrix_36077725286422
// MI455X (gfx1250) — matrix operand order repaired and checked by reading, not compiled, not run on hardware
//
#include <hip/hip_runtime.h>
#include <hip/hip_bf16.h>

// ---------- types ----------
typedef __attribute__((ext_vector_type(16))) _Float16 v16h;
typedef __attribute__((ext_vector_type(8)))  _Float16 v8h;
typedef __attribute__((ext_vector_type(4)))  _Float16 v4h;
typedef __attribute__((ext_vector_type(8)))  float    v8f;
typedef __attribute__((ext_vector_type(4)))  float    v4f;

#define BQ 4
#define NQ 256
#define DQ 512
#define MQ (BQ * NQ)   // 1024 rows for the [B*N, D] GEMMs
#define ZSTR 520       // z-tile LDS row stride in halves (512 + 8 pad -> conflict-free ds_load_b128)

// ---------- helpers ----------
static __device__ __forceinline__ v16h cat16(v8h lo, v8h hi) {
  return __builtin_shufflevector(lo, hi, 0,1,2,3,4,5,6,7,8,9,10,11,12,13,14,15);
}

// 16 contiguous fp32 -> v16h  (B-fragment: lane covers K = half*16 + 0..15)
static __device__ __forceinline__ v16h cvt16(const float* __restrict__ p) {
  v4f f0 = *(const v4f*)(p);
  v4f f1 = *(const v4f*)(p + 4);
  v4f f2 = *(const v4f*)(p + 8);
  v4f f3 = *(const v4f*)(p + 12);
  v16h r;
  r[0]=(_Float16)f0[0];  r[1]=(_Float16)f0[1];  r[2]=(_Float16)f0[2];  r[3]=(_Float16)f0[3];
  r[4]=(_Float16)f1[0];  r[5]=(_Float16)f1[1];  r[6]=(_Float16)f1[2];  r[7]=(_Float16)f1[3];
  r[8]=(_Float16)f2[0];  r[9]=(_Float16)f2[1];  r[10]=(_Float16)f2[2]; r[11]=(_Float16)f2[3];
  r[12]=(_Float16)f3[0]; r[13]=(_Float16)f3[1]; r[14]=(_Float16)f3[2]; r[15]=(_Float16)f3[3];
  return r;
}

// A-fragment from fp32 row: elems 0..15 <- p[0..15]
// (p already offset by kc*32 + half*16)
static __device__ __forceinline__ v16h cvtA(const float* __restrict__ p) {
  v4f f0 = *(const v4f*)(p);
  v4f f1 = *(const v4f*)(p + 4);
  v4f f2 = *(const v4f*)(p + 8);
  v4f f3 = *(const v4f*)(p + 12);
  v16h r;
  r[0]=(_Float16)f0[0];  r[1]=(_Float16)f0[1];  r[2]=(_Float16)f0[2];  r[3]=(_Float16)f0[3];
  r[4]=(_Float16)f1[0];  r[5]=(_Float16)f1[1];  r[6]=(_Float16)f1[2];  r[7]=(_Float16)f1[3];
  r[8]=(_Float16)f2[0];  r[9]=(_Float16)f2[1];  r[10]=(_Float16)f2[2]; r[11]=(_Float16)f2[3];
  r[12]=(_Float16)f3[0]; r[13]=(_Float16)f3[1]; r[14]=(_Float16)f3[2]; r[15]=(_Float16)f3[3];
  return r;
}

static __device__ __forceinline__ v8f wmma16(v16h a, v16h b, v8f c) {
  return __builtin_amdgcn_wmma_f32_16x16x32_f16(false, a, false, b, (short)0, c, false, false);
}

static __device__ __forceinline__ float sigf(float x) {
  return 1.0f / (1.0f + __expf(-x));
}

// ---------- kernel 1: one highway layer (launched 3x) ----------
// y = g * relu(x@tw^T + tb) + (1-g) * x ;  g = sigmoid(x@gw^T + gb)
// grid 256 blocks * 256 threads = 2048 waves; wave tile = (mt in 64) x (et in 32)
__global__ __launch_bounds__(256) void hw_layer_k(
    const float* __restrict__ x, const float* __restrict__ gw, const float* __restrict__ gb,
    const float* __restrict__ tw, const float* __restrict__ tb, float* __restrict__ y) {
  const int wv   = threadIdx.x >> 5;
  const int lane = threadIdx.x & 31;
  const int hh   = lane >> 4;       // lane half
  const int ln   = lane & 15;
  const int tile = blockIdx.x * 8 + wv;
  const int mt = tile >> 5, et = tile & 31;

  const float* arow = x  + (size_t)(mt * 16 + ln) * DQ;
  const float* grow = gw + (size_t)(et * 16 + ln) * DQ + hh * 16;
  const float* trow = tw + (size_t)(et * 16 + ln) * DQ + hh * 16;

  v8f cg = {}; v8f ct = {};
  #pragma unroll 4
  for (int kc = 0; kc < 16; ++kc) {
    const int kb = kc * 32;
    v16h af = cvtA(arow + kb + hh * 16);
    v16h bg = cvt16(grow + kb);
    v16h bt = cvt16(trow + kb);
    cg = wmma16(af, bg, cg);
    ct = wmma16(af, bt, ct);
  }

  const int e = et * 16 + ln;
  const float gbias = gb[e], tbias = tb[e];
  #pragma unroll
  for (int r = 0; r < 8; ++r) {
    const int row = mt * 16 + hh * 8 + r;
    const float xv = x[(size_t)row * DQ + e];
    const float g  = sigf(cg[r] + gbias);
    const float t  = fmaxf(ct[r] + tbias, 0.0f);
    y[(size_t)row * DQ + e] = g * t + (1.0f - g) * xv;
  }
}

// ---------- kernel 2: a = h@W0[:, :D]^T + b0 ; bb = h@W0[:, D:]^T ----------
__global__ __launch_bounds__(256) void ab_k(
    const float* __restrict__ h, const float* __restrict__ lw0w, const float* __restrict__ lw0b,
    float* __restrict__ aout, float* __restrict__ bout) {
  const int wv   = threadIdx.x >> 5;
  const int lane = threadIdx.x & 31;
  const int hh   = lane >> 4;
  const int ln   = lane & 15;
  const int tile = blockIdx.x * 8 + wv;
  const int mt = tile >> 5, et = tile & 31;

  const float* arow = h + (size_t)(mt * 16 + ln) * DQ;
  const float* w0   = lw0w + (size_t)(et * 16 + ln) * (2 * DQ) + hh * 16;   // first-half cols
  const float* w1   = w0 + DQ;                                              // second-half cols

  v8f ca = {}; v8f cb = {};
  #pragma unroll 4
  for (int kc = 0; kc < 16; ++kc) {
    const int kb = kc * 32;
    v16h af = cvtA(arow + kb + hh * 16);
    v16h b0 = cvt16(w0 + kb);
    v16h b1 = cvt16(w1 + kb);
    ca = wmma16(af, b0, ca);
    cb = wmma16(af, b1, cb);
  }

  const int e = et * 16 + ln;
  const float abias = lw0b[e];
  #pragma unroll
  for (int r = 0; r < 8; ++r) {
    const int row = mt * 16 + hh * 8 + r;
    aout[(size_t)row * DQ + e] = ca[r] + abias;   // lw0_b folded into a
    bout[(size_t)row * DQ + e] = cb[r];
  }
}

// ---------- kernel 3: convert lw1_w fp32 -> fp16 (stays hot in 192MB L2) ----------
__global__ __launch_bounds__(256) void cvt_lw1_k(const float* __restrict__ w, _Float16* __restrict__ o) {
  const int idx = (blockIdx.x * 256 + threadIdx.x) * 4;
  v4f f = *(const v4f*)(w + idx);
  v4h hv;
  hv[0]=(_Float16)f[0]; hv[1]=(_Float16)f[1]; hv[2]=(_Float16)f[2]; hv[3]=(_Float16)f[3];
  *(v4h*)(o + idx) = hv;
}

// ---------- kernel 4: fused pairwise z-GEMM + lwo reduction + sigmoid ----------
// One block per (b, 16x16 i-tile, 16x16 j-tile). z tile (256 pairs x 512 K, fp16)
// lives entirely in LDS (260KB of the 320KB WGP LDS) — never touches HBM.
// Register blocking: 2 M-tiles x 4 e-tiles per wave pass -> 8 WMMAs per
// (2 ds_b128-pair A loads + 4 global_b128-pair B loads) = 1.5 loads/WMMA.
__global__ __launch_bounds__(256) void pair_k(
    const float* __restrict__ a, const float* __restrict__ bb,
    const _Float16* __restrict__ lw1h, const float* __restrict__ lw1b,
    const float* __restrict__ lwow, const float* __restrict__ lwob,
    float* __restrict__ out) {
  extern __shared__ char smem[];
  _Float16* zsh   = (_Float16*)smem;                        // 256 * ZSTR halves
  float*   s_l1b  = (float*)(smem + 256 * ZSTR * 2);        // 512
  float*   s_lwo  = s_l1b + DQ;                             // 512

  const int t   = threadIdx.x;
  const int blk = blockIdx.x;
  const int b   = blk >> 8;
  const int rem = blk & 255;
  const int i0  = (rem >> 4) * 16;
  const int j0  = (rem & 15) * 16;

  // preload biases / output weights
  s_l1b[t]       = lw1b[t];
  s_l1b[t + 256] = lw1b[t + 256];
  s_lwo[t]       = lwow[t];
  s_lwo[t + 256] = lwow[t + 256];

  // build z tile: thread t owns pair p=t (i=p>>4, j=p&15), all 512 K
  {
    const int i = t >> 4, j = t & 15;
    const float* ar = a  + (size_t)(b * NQ + i0 + i) * DQ;
    const float* br = bb + (size_t)(b * NQ + j0 + j) * DQ;
    _Float16* zr = zsh + t * ZSTR;
    #pragma unroll 4
    for (int k = 0; k < DQ; k += 4) {
      v4f av = *(const v4f*)(ar + k);
      v4f bv = *(const v4f*)(br + k);
      v4h zp;
      zp[0] = (_Float16)fmaxf(av[0] + bv[0], 0.0f);
      zp[1] = (_Float16)fmaxf(av[1] + bv[1], 0.0f);
      zp[2] = (_Float16)fmaxf(av[2] + bv[2], 0.0f);
      zp[3] = (_Float16)fmaxf(av[3] + bv[3], 0.0f);
      *(v4h*)(zr + k) = zp;
    }
  }
  __syncthreads();

  const int wv   = t >> 5;
  const int lane = t & 31;
  const int hh   = lane >> 4;
  const int ln   = lane & 15;
  const int m0 = 2 * wv, m1 = 2 * wv + 1;      // two 16-row M tiles per wave
  const int rowoff0 = (m0 * 16 + ln) * ZSTR;
  const int rowoff1 = (m1 * 16 + ln) * ZSTR;

  float r0[8], r1[8];
  #pragma unroll
  for (int r = 0; r < 8; ++r) { r0[r] = 0.0f; r1[r] = 0.0f; }

  // outer: 8 groups of 4 e-tiles; inner: K sweep with 8 live accumulators
  for (int eo = 0; eo < 8; ++eo) {
    v8f c0[4], c1[4];
    #pragma unroll
    for (int ei = 0; ei < 4; ++ei) { c0[ei] = (v8f){}; c1[ei] = (v8f){}; }

    const _Float16* bbs[4];
    #pragma unroll
    for (int ei = 0; ei < 4; ++ei)
      bbs[ei] = lw1h + (size_t)(((eo * 4 + ei) * 16) + ln) * DQ + hh * 16;

    #pragma unroll 2
    for (int kc = 0; kc < 16; ++kc) {
      const int kb = kc * 32;
      const int ab = kb + hh * 16;
      v16h a0 = cat16(*(const v8h*)(zsh + rowoff0 + ab), *(const v8h*)(zsh + rowoff0 + ab + 8));
      v16h a1 = cat16(*(const v8h*)(zsh + rowoff1 + ab), *(const v8h*)(zsh + rowoff1 + ab + 8));
      #pragma unroll
      for (int ei = 0; ei < 4; ++ei) {
        const _Float16* bp = bbs[ei] + kb;
        v16h bf = cat16(*(const v8h*)(bp), *(const v8h*)(bp + 8));
        c0[ei] = wmma16(a0, bf, c0[ei]);
        c1[ei] = wmma16(a1, bf, c1[ei]);
      }
    }

    #pragma unroll
    for (int ei = 0; ei < 4; ++ei) {
      const int e = (eo * 4 + ei) * 16 + ln;
      const float wo  = s_lwo[e];
      const float b1b = s_l1b[e];
      #pragma unroll
      for (int r = 0; r < 8; ++r) {
        r0[r] += wo * fmaxf(c0[ei][r] + b1b, 0.0f);
        r1[r] += wo * fmaxf(c1[ei][r] + b1b, 0.0f);
      }
    }
  }

  // reduce over the 16 e-columns held by the 16 lanes of each half
  #pragma unroll
  for (int m = 1; m < 16; m <<= 1) {
    #pragma unroll
    for (int r = 0; r < 8; ++r) {
      r0[r] += __shfl_xor(r0[r], m);
      r1[r] += __shfl_xor(r1[r], m);
    }
  }

  const float ob = lwob[0];
  if (ln == 0) {
    #pragma unroll
    for (int r = 0; r < 8; ++r) {
      int p = m0 * 16 + hh * 8 + r;
      out[(size_t)(b * NQ + i0 + (p >> 4)) * NQ + j0 + (p & 15)] = sigf(r0[r] + ob);
      p = m1 * 16 + hh * 8 + r;
      out[(size_t)(b * NQ + i0 + (p >> 4)) * NQ + j0 + (p & 15)] = sigf(r1[r] + ob);
    }
  }
}

// ---------- launch ----------
extern "C" void kernel_launch(void* const* d_in, const int* in_sizes, int n_in,
                              void* d_out, int out_size, void* d_ws, size_t ws_size,
                              hipStream_t stream) {
  (void)in_sizes; (void)n_in; (void)out_size; (void)ws_size;
  const float* x    = (const float*)d_in[0];
  const float* gw   = (const float*)d_in[1];
  const float* gb   = (const float*)d_in[2];
  const float* tw   = (const float*)d_in[3];
  const float* tb   = (const float*)d_in[4];
  const float* lw0w = (const float*)d_in[5];
  const float* lw0b = (const float*)d_in[6];
  const float* lw1w = (const float*)d_in[7];
  const float* lw1b = (const float*)d_in[8];
  const float* lwow = (const float*)d_in[9];
  const float* lwob = (const float*)d_in[10];
  float* out = (float*)d_out;

  float* h0 = (float*)d_ws;                 // 1024*512 f32 = 2MB
  float* h1 = h0 + (size_t)MQ * DQ;         // 2MB
  float* aw = h1 + (size_t)MQ * DQ;         // 2MB
  float* bw = aw + (size_t)MQ * DQ;         // 2MB
  _Float16* lw1h = (_Float16*)(bw + (size_t)MQ * DQ);  // 512KB

  const size_t WD = (size_t)DQ * DQ;        // per-layer weight stride

  // 3 highway layers (ping-pong h0/h1)
  hw_layer_k<<<256, 256, 0, stream>>>(x,  gw,          gb,        tw,          tb,        h0);
  hw_layer_k<<<256, 256, 0, stream>>>(h0, gw + WD,     gb + DQ,   tw + WD,     tb + DQ,   h1);
  hw_layer_k<<<256, 256, 0, stream>>>(h1, gw + 2 * WD, gb + 2*DQ, tw + 2 * WD, tb + 2*DQ, h0);

  // a/b projections (+ lw0_b folded into a) and fp16 weight conversion
  cvt_lw1_k<<<256, 256, 0, stream>>>(lw1w, lw1h);
  ab_k<<<256, 256, 0, stream>>>(h0, lw0w, lw0b, aw, bw);

  // fused pairwise kernel: 260KB z tile + 4KB bias/weight tables in LDS
  const size_t smem_bytes = (size_t)256 * ZSTR * 2 + 2 * DQ * 4;
  pair_k<<<1024, 256, smem_bytes, stream>>>(aw, bw, lw1h, lw1b, lwow, lwob, out);
}
